// FAVORplus_4028679324110
// MI455X (gfx1250) — matrix operand order repaired and checked by reading, not compiled, not run on hardware
//
#include <hip/hip_runtime.h>

// ---------------------------------------------------------------------------
// FAVOR+ attention, fused for MI455X (gfx1250, wave32, bf16 WMMA).
//   B=8 H=4 L=4096 D=E=128 M=640, fp32 I/O.
// Roofline: ~86 GFLOP matrix math vs 256 MB mandatory HBM traffic (~11us at
// 23.3 TB/s). We fuse so phi (2x320MB) never touches HBM; only a 5MB bf16
// context + 80KB ksum go through the workspace (L2-resident, 192MB L2).
// Matrix math uses v_wmma_f32_16x16x32_bf16 (fp32 accumulate).
// ---------------------------------------------------------------------------

#define LSEQ 4096
#define DH   128
#define MF   640

#define EPS_K      1e-4f
#define INV_SQSQ_D 0.2973017787506803f   // 128^-0.25 (folded into projection)
#define INV_SQ_M   0.03952847075210474f  // 640^-0.5
#define INV_2SQD   0.04419417382415922f  // 1/(2*sqrt(128))

typedef __attribute__((ext_vector_type(16))) __bf16 v16bf;
typedef __attribute__((ext_vector_type(8)))  float  v8f;

__device__ __forceinline__ unsigned short f2bf(float f) {
  unsigned int u = __float_as_uint(f);
  u += 0x7FFFu + ((u >> 16) & 1u);   // round-to-nearest-even
  return (unsigned short)(u >> 16);
}

// 16-element bf16 fragment from two contiguous 16B regions (-> ds_load_b128 x2)
__device__ __forceinline__ v16bf ld_frag(const unsigned short* p0,
                                         const unsigned short* p1) {
  v16bf v;
  __builtin_memcpy(&v, p0, 16);
  __builtin_memcpy(((char*)&v) + 16, p1, 16);
  return v;
}

// --------------------------- K0: projection ---> bf16 ----------------------
__global__ __launch_bounds__(256)
void convert_proj(const float* __restrict__ proj, unsigned short* __restrict__ Pbf) {
  int i = blockIdx.x * 256 + threadIdx.x;           // 640*128 = 81920
  if (i < MF * DH) Pbf[i] = f2bf(proj[i] * INV_SQSQ_D);
}

// ------------- K1: phi_k, ksum, context = phi_k^T @ V (per m-chunk) --------
// grid (5 m-chunks, 32 heads), 256 threads = 8 waves. Wave w owns m-tile w
// (16 feature columns) of its 128-wide chunk; context accumulators (8 E-tiles
// x 8 VGPR) live in registers for the whole L loop.
__global__ __launch_bounds__(256)
void k1_phik_ctx(const float* __restrict__ kg, const float* __restrict__ vg,
                 const unsigned short* __restrict__ Pbf,
                 unsigned short* __restrict__ ctxg, float* __restrict__ ksumg) {
  const int m0   = blockIdx.x * 128;
  const int bh   = blockIdx.y;
  const int tid  = threadIdx.x;
  const int wave = tid >> 5;
  const int lane = tid & 31;
  const int half = lane >> 4;
  const int lr   = lane & 15;

  __shared__ unsigned short Plds[128 * 128];  // P' chunk  [m][d]   32KB
  __shared__ unsigned short kbf [32 * 128];   // k tile    [row][d]  8KB
  __shared__ unsigned short vT  [128 * 32];   // v tile^T  [e][row]  8KB
  __shared__ unsigned short phiT[128 * 32];   // phi^T     [m][row]  8KB
  __shared__ float gLds[32];

  { // load projection chunk once (B-operand, [m][d] == transposed P'^T)
    const uint4* ps = (const uint4*)(Pbf + (size_t)m0 * DH);
    uint4* pd = (uint4*)Plds;
#pragma unroll
    for (int i = 0; i < 8; ++i) pd[tid + i * 256] = ps[tid + i * 256];
  }

  v8f vzero = {};
  v8f ctxacc[8];
#pragma unroll
  for (int e = 0; e < 8; ++e) ctxacc[e] = vzero;
  float ksum_acc = 0.0f;

  const float* kbase = kg + (size_t)bh * LSEQ * DH;
  const float* vbase = vg + (size_t)bh * LSEQ * DH;

  for (int r0 = 0; r0 < LSEQ; r0 += 32) {
    __syncthreads();                       // prev iter done reading kbf/vT
    { // ---- stage k (bf16, row-major) and v (bf16, transposed) tiles ------
      const float4* k4 = (const float4*)(kbase + (size_t)r0 * DH);
      const float4* v4 = (const float4*)(vbase + (size_t)r0 * DH);
      unsigned int* kbp = (unsigned int*)kbf;
#pragma unroll
      for (int it = 0; it < 4; ++it) {
        int fi = tid + it * 256;
        int row = fi >> 5, cg = fi & 31;
        float4 kf = k4[fi];
        kbp[row * 64 + cg * 2]     = (unsigned)f2bf(kf.x) | ((unsigned)f2bf(kf.y) << 16);
        kbp[row * 64 + cg * 2 + 1] = (unsigned)f2bf(kf.z) | ((unsigned)f2bf(kf.w) << 16);
        float4 vf = v4[fi];
        vT[(cg * 4 + 0) * 32 + row] = f2bf(vf.x);
        vT[(cg * 4 + 1) * 32 + row] = f2bf(vf.y);
        vT[(cg * 4 + 2) * 32 + row] = f2bf(vf.z);
        vT[(cg * 4 + 3) * 32 + row] = f2bf(vf.w);
      }
      if (tid < 32) {                      // deterministic row norms (f32)
        const float4* kr = (const float4*)(kbase + (size_t)(r0 + tid) * DH);
        float s = 0.0f;
#pragma unroll
        for (int c = 0; c < 32; ++c) {
          float4 f = kr[c];
          s += f.x * f.x + f.y * f.y + f.z * f.z + f.w * f.w;
        }
        gLds[tid] = s * INV_2SQD;
      }
    }
    __syncthreads();

    // ---- phase A: phi_k tiles [32 rows x 16 cols(mt=wave)] --------------
    v16bf Bf[4];
#pragma unroll
    for (int kk = 0; kk < 4; ++kk) {       // B frag: N=m, K=d (table order)
      const unsigned short* p = &Plds[(wave * 16 + lr) * 128 + kk * 32 + half * 8];
      Bf[kk] = ld_frag(p, p + 16);
    }
#pragma unroll
    for (int rs = 0; rs < 2; ++rs) {
      v8f acc = vzero;
#pragma unroll
      for (int kk = 0; kk < 4; ++kk) {     // A frag: M=row, K=d
        const unsigned short* a0 = &kbf[(rs * 16 + lr) * 128 + kk * 32 + half * 8];
        v16bf Af = ld_frag(a0, a0 + 16);
        acc = __builtin_amdgcn_wmma_f32_16x16x32_bf16(false, Af, false, Bf[kk],
                                                      (short)0, acc, false, false);
      }
      unsigned short pk[8];
#pragma unroll
      for (int vv = 0; vv < 8; ++vv) {
        int rloc = rs * 16 + half * 8 + vv;
        float phi = __expf(acc[vv] - gLds[rloc] + EPS_K) * INV_SQ_M;
        ksum_acc += phi;                   // column = wave*16+lr (fixed/lane)
        pk[vv] = f2bf(phi);
      }
      __builtin_memcpy(&phiT[(wave * 16 + lr) * 32 + rs * 16 + half * 8], pk, 16);
    }
    __syncthreads();

    // ---- phase B: ctx[m,e] += phi^T @ v  (K = 32 rows) ------------------
    const unsigned short* ap = &phiT[(wave * 16 + lr) * 32 + half * 8];
    v16bf Aphi = ld_frag(ap, ap + 16);
#pragma unroll
    for (int et = 0; et < 8; ++et) {
      const unsigned short* bp = &vT[(et * 16 + lr) * 32 + half * 8];
      v16bf Bv = ld_frag(bp, bp + 16);
      ctxacc[et] = __builtin_amdgcn_wmma_f32_16x16x32_bf16(false, Aphi, false, Bv,
                                                           (short)0, ctxacc[et],
                                                           false, false);
    }
  }

  { // ksum: combine row halves (lane ^ 16 holds the other 16 rows)
    float val = ksum_acc + __shfl_xor(ksum_acc, 16);
    if (lane < 16)
      ksumg[(size_t)bh * MF + m0 + wave * 16 + lane] = val;
  }
#pragma unroll
  for (int et = 0; et < 8; ++et) {         // ctx chunk -> bf16 workspace
#pragma unroll
    for (int vv = 0; vv < 8; ++vv) {
      int mfeat = m0 + wave * 16 + half * 8 + vv;
      ctxg[((size_t)bh * MF + mfeat) * DH + et * 16 + lr] = f2bf(ctxacc[et][vv]);
    }
  }
}

// --------- K2: phi_q, denominator, out = (phi_q @ C) / (phi_q . S) ---------
// grid (32 row-chunks, 32 heads). Wave w owns 16 rows; out accumulators
// (8 E-tiles) in registers; M swept in 64-wide chunks through LDS.
__global__ __launch_bounds__(256)
void k2_phiq_out(const float* __restrict__ qg,
                 const unsigned short* __restrict__ Pbf,
                 const unsigned short* __restrict__ ctxg,
                 const float* __restrict__ ksumg,
                 float* __restrict__ outg) {
  const int r0   = blockIdx.x * 128;
  const int bh   = blockIdx.y;
  const int tid  = threadIdx.x;
  const int wave = tid >> 5, lane = tid & 31;
  const int half = lane >> 4, lr = lane & 15;

  __shared__ unsigned short Pl  [64 * 128];   // P' m-chunk [m][d]    16KB
  __shared__ unsigned short CbT [128 * 64];   // ctx^T      [e][m]    16KB
  __shared__ unsigned short phiB[128 * 64];   // phi_q      [row][m]  16KB
  __shared__ float ksumL[MF];
  __shared__ float gq[128];

  const float* qbase = qg + ((size_t)bh * LSEQ + r0) * DH;

  for (int i = tid; i < MF; i += 256) ksumL[i] = ksumg[(size_t)bh * MF + i];
  if (tid < 128) {                           // deterministic q row norms
    const float4* qr = (const float4*)(qbase + (size_t)tid * DH);
    float s = 0.0f;
#pragma unroll
    for (int c = 0; c < 32; ++c) {
      float4 f = qr[c];
      s += f.x * f.x + f.y * f.y + f.z * f.z + f.w * f.w;
    }
    gq[tid] = s * INV_2SQD;
  }

  // q A-fragments straight from global f32 (rows wave*16+lr, K=0..127)
  v16bf Aq[4];
  {
    const float* rowp = qbase + (size_t)(wave * 16 + lr) * DH;
#pragma unroll
    for (int kk = 0; kk < 4; ++kk) {
      float4 a0 = *(const float4*)(rowp + kk * 32 + half * 8);
      float4 a1 = *(const float4*)(rowp + kk * 32 + half * 8 + 4);
      float4 b0 = *(const float4*)(rowp + kk * 32 + 16 + half * 8);
      float4 b1 = *(const float4*)(rowp + kk * 32 + 16 + half * 8 + 4);
      unsigned short u[16] = {
        f2bf(a0.x), f2bf(a0.y), f2bf(a0.z), f2bf(a0.w),
        f2bf(a1.x), f2bf(a1.y), f2bf(a1.z), f2bf(a1.w),
        f2bf(b0.x), f2bf(b0.y), f2bf(b0.z), f2bf(b0.w),
        f2bf(b1.x), f2bf(b1.y), f2bf(b1.z), f2bf(b1.w) };
      __builtin_memcpy(&Aq[kk], u, 32);
    }
  }

  v8f vzero = {};
  v8f outacc[8];
#pragma unroll
  for (int e = 0; e < 8; ++e) outacc[e] = vzero;
  float dpart[8];
#pragma unroll
  for (int vv = 0; vv < 8; ++vv) dpart[vv] = 0.0f;

  for (int mc = 0; mc < 10; ++mc) {
    const int mk0 = mc * 64;
    __syncthreads();                       // prev iter done reading Pl/CbT
    {
      const uint4* ps = (const uint4*)(Pbf + (size_t)mk0 * DH);
      uint4* pd = (uint4*)Pl;
#pragma unroll
      for (int i = 0; i < 4; ++i) pd[tid + i * 256] = ps[tid + i * 256];
    }
    { // ctx chunk -> LDS transposed [e][m]
      const uint4* cs = (const uint4*)(ctxg + ((size_t)bh * MF + mk0) * DH);
#pragma unroll
      for (int i = 0; i < 4; ++i) {
        int fi = tid + i * 256;
        int ml = fi >> 4, eg = fi & 15;
        uint4 t = cs[fi];
        unsigned short u[8];
        __builtin_memcpy(u, &t, 16);
#pragma unroll
        for (int c = 0; c < 8; ++c) CbT[(eg * 8 + c) * 64 + ml] = u[c];
      }
    }
    __syncthreads();

    // phi_q tiles for this m-chunk (+ denominator partials)
#pragma unroll
    for (int mt = 0; mt < 4; ++mt) {
      v8f acc = vzero;
#pragma unroll
      for (int kk = 0; kk < 4; ++kk) {
        const unsigned short* bp = &Pl[(mt * 16 + lr) * 128 + kk * 32 + half * 8];
        v16bf Bf = ld_frag(bp, bp + 16);
        acc = __builtin_amdgcn_wmma_f32_16x16x32_bf16(false, Aq[kk], false, Bf,
                                                      (short)0, acc, false, false);
      }
      float ks = ksumL[mk0 + mt * 16 + lr];
#pragma unroll
      for (int vv = 0; vv < 8; ++vv) {
        int rloc = wave * 16 + half * 8 + vv;
        float phi = __expf(acc[vv] - gq[rloc] + EPS_K) * INV_SQ_M;
        dpart[vv] += phi * ks;
        phiB[rloc * 64 + mt * 16 + lr] = f2bf(phi);
      }
    }
    // same-wave store->load through LDS: drain DS counter (CDNA5 split wait)
    asm volatile("s_wait_dscnt 0" ::: "memory");

    // out += phi_q @ C over this 64-wide K slab
#pragma unroll
    for (int kp = 0; kp < 2; ++kp) {
      const unsigned short* ap = &phiB[(wave * 16 + lr) * 64 + kp * 32 + half * 8];
      v16bf Af = ld_frag(ap, ap + 16);
#pragma unroll
      for (int et = 0; et < 8; ++et) {
        const unsigned short* bp = &CbT[(et * 16 + lr) * 64 + kp * 32 + half * 8];
        v16bf Bf = ld_frag(bp, bp + 16);
        outacc[et] = __builtin_amdgcn_wmma_f32_16x16x32_bf16(false, Af, false, Bf,
                                                             (short)0, outacc[et],
                                                             false, false);
      }
    }
  }

  // denominator: reduce across the 16 lanes of each half (rows fixed per lane)
#pragma unroll
  for (int vv = 0; vv < 8; ++vv) {
    float d = dpart[vv];
    d += __shfl_xor(d, 1);
    d += __shfl_xor(d, 2);
    d += __shfl_xor(d, 4);
    d += __shfl_xor(d, 8);
    dpart[vv] = d;
  }
  float* obase = outg + ((size_t)bh * LSEQ + r0) * DH;
#pragma unroll
  for (int et = 0; et < 8; ++et) {
#pragma unroll
    for (int vv = 0; vv < 8; ++vv) {
      int row = wave * 16 + half * 8 + vv;
      obase[(size_t)row * DH + et * 16 + lr] = outacc[et][vv] / dpart[vv];
    }
  }
}

// ------------------------------- launcher ----------------------------------
extern "C" void kernel_launch(void* const* d_in, const int* in_sizes, int n_in,
                              void* d_out, int out_size, void* d_ws, size_t ws_size,
                              hipStream_t stream) {
  const float* q    = (const float*)d_in[0];
  const float* k    = (const float*)d_in[1];
  const float* v    = (const float*)d_in[2];
  const float* proj = (const float*)d_in[3];
  float* out = (float*)d_out;

  char* ws = (char*)d_ws;
  unsigned short* Pbf   = (unsigned short*)ws;                 // 640*128*2   = 160KB
  float*          ksumg = (float*)(ws + 163840);               // 32*640*4    =  80KB
  unsigned short* ctxg  = (unsigned short*)(ws + 245760);      // 32*640*128*2=  5MB

  convert_proj<<<dim3(320), dim3(256), 0, stream>>>(proj, Pbf);
  k1_phik_ctx<<<dim3(5, 32), dim3(256), 0, stream>>>(k, v, Pbf, ctxg, ksumg);
  k2_phiq_out<<<dim3(32, 32), dim3(256), 0, stream>>>(q, Pbf, ctxg, ksumg, out);
}
